// Attention_48361331753083
// MI455X (gfx1250) — matrix operand order repaired and checked by reading, not compiled, not run on hardware
//
#include <hip/hip_runtime.h>

// ---------------------------------------------------------------------------
// Types / helpers
// ---------------------------------------------------------------------------
typedef __attribute__((ext_vector_type(16))) unsigned short ushort16v;
typedef __attribute__((ext_vector_type(16))) __bf16        bf16x16;
typedef __attribute__((ext_vector_type(8)))  float         floatx8;

#if defined(__has_builtin)
#  if __has_builtin(__builtin_amdgcn_tensor_load_to_lds)
#    define HAVE_TDM 1
#  endif
#endif
#ifndef HAVE_TDM
#  define HAVE_TDM 0
#endif

static __device__ __forceinline__ unsigned short f2bf(float f) {
  union { float f; unsigned int u; } cv; cv.f = f;
  unsigned int u = cv.u;
  u += 0x7FFFu + ((u >> 16) & 1u);            // round-to-nearest-even
  return (unsigned short)(u >> 16);
}

static __device__ __forceinline__ floatx8 wmma_bf16(ushort16v a, ushort16v b, floatx8 c) {
  return __builtin_amdgcn_wmma_f32_16x16x32_bf16(
      /*neg_a=*/false, __builtin_bit_cast(bf16x16, a),
      /*neg_b=*/false, __builtin_bit_cast(bf16x16, b),
      /*c_mod=*/(short)0, c, /*reuse_a=*/false, /*reuse_b=*/false);
}

#if HAVE_TDM
typedef __attribute__((ext_vector_type(4))) unsigned int uint32x4_t_;
typedef __attribute__((ext_vector_type(8))) int          int32x8_t_;
typedef __attribute__((ext_vector_type(4))) int          int32x4_t_;

// 2-D bf16 tile DMA: global (row-major, row_stride elems) -> LDS, with the
// TDM pad feature producing the padded LDS row stride our fragment reads use.
// D# layout per CDNA5 ISA (group0: count/lds/global/type,
// group1: data_size|pad, tensor dims, tile dims, dim0 stride).
static __device__ __forceinline__ void tdm_load_2d(
    unsigned lds_off, const void* gptr,
    unsigned width_elems, unsigned height_rows,
    unsigned long long row_stride_elems,
    unsigned pad_interval_code, unsigned pad_amount_code) {
  unsigned long long ga = (unsigned long long)gptr;
  uint32x4_t_ g0;
  g0[0] = 1u;                                            // count=1 (valid user D#)
  g0[1] = lds_off;                                       // LDS byte address
  g0[2] = (unsigned)(ga & 0xFFFFFFFFu);                  // global_addr lo
  g0[3] = (unsigned)((ga >> 32) & 0x01FFFFFFu) | (2u << 30);  // addr hi | type=2
  int32x8_t_ g1;
  g1[0] = (int)((1u << 16)                               // data_size=1 -> 2 bytes
              | (1u << 20)                               // pad_enable
              | (pad_interval_code << 22)
              | (pad_amount_code << 25));
  g1[1] = (int)((width_elems & 0xFFFFu) << 16);          // tensor_dim0 lo16
  g1[2] = (int)(((width_elems >> 16) & 0xFFFFu)          // tensor_dim0 hi16
              | ((height_rows & 0xFFFFu) << 16));        // tensor_dim1 lo16
  g1[3] = (int)(((height_rows >> 16) & 0xFFFFu)          // tensor_dim1 hi16
              | ((width_elems & 0xFFFFu) << 16));        // tile_dim0
  g1[4] = (int)(height_rows & 0xFFFFu);                  // tile_dim1 (tile_dim2=0)
  g1[5] = (int)(row_stride_elems & 0xFFFFFFFFull);       // tensor_dim0_stride lo
  g1[6] = (int)((row_stride_elems >> 32) & 0xFFFFull);   // stride hi16
  g1[7] = 0;
  int32x4_t_ gz = {0, 0, 0, 0};
#if __clang_major__ >= 23
  int32x8_t_ gz8 = {0, 0, 0, 0, 0, 0, 0, 0};
  __builtin_amdgcn_tensor_load_to_lds(g0, g1, gz, gz, gz8, 0);
#else
  __builtin_amdgcn_tensor_load_to_lds(g0, g1, gz, gz, 0);
#endif
}
#endif  // HAVE_TDM

// ---------------------------------------------------------------------------
// fp32 -> bf16 rounding pass
// ---------------------------------------------------------------------------
__global__ void __launch_bounds__(256) f32_to_bf16_kernel(
    const float* __restrict__ in, unsigned short* __restrict__ out, int n) {
  int i = blockIdx.x * 256 + threadIdx.x;
  if (i < n) out[i] = f2bf(in[i]);
}

// ---------------------------------------------------------------------------
// bf16 GEMM: C[M,N] = alpha * A[M,K] @ B[K,N]   (row-major, f32 accumulate)
// Block tile 128x128, K-step 32, 8 waves: wave = 32x64 of C = 2x4 WMMA accs.
// A tile staged by TDM (pad-> stride 40 u16); B tile stored transposed [N][K].
// ---------------------------------------------------------------------------
template <bool STORE_BF16>
__global__ void __launch_bounds__(256) gemm_tile(
    const unsigned short* __restrict__ A,
    const unsigned short* __restrict__ Bm,
    void* __restrict__ Cout,
    int M, int N, int K, float alpha) {
  __shared__ unsigned short As[128 * 40];   // 64B row + 16B pad (TDM pad 3/3)
  __shared__ unsigned short Bs[128 * 40];

  const int tid  = threadIdx.x;
  const int lane = tid & 31;
  const int w    = tid >> 5;
  const int wm   = w & 3;        // 4 waves along M: 32 rows each
  const int wn   = w >> 2;       // 2 waves along N: 64 cols each
  const int half = lane >> 4;
  const int l15  = lane & 15;
  const int mblk = blockIdx.y * 128;
  const int nblk = blockIdx.x * 128;

  floatx8 acc[2][4];
#pragma unroll
  for (int i = 0; i < 2; ++i)
#pragma unroll
    for (int j = 0; j < 4; ++j)
#pragma unroll
      for (int r = 0; r < 8; ++r) acc[i][j][r] = 0.0f;

  for (int kt = 0; kt < K; kt += 32) {
    __syncthreads();   // previous tile fully consumed
#if HAVE_TDM
    if (w == 0)        // one wave drives the DMA: 128x32 bf16, stride K
      tdm_load_2d((unsigned)(uintptr_t)As, A + (size_t)mblk * K + kt,
                  /*w=*/32, /*h=*/128, (unsigned long long)K,
                  /*interval 8x8B=64B*/ 3, /*pad 4 dw=16B*/ 3);
#else
#pragma unroll
    for (int i = 0; i < 2; ++i) {
      int v   = tid + i * 256;
      int row = v >> 2;
      int c8  = (v & 3) * 8;
      *(uint4*)(&As[row * 40 + c8]) =
          *(const uint4*)(A + (size_t)(mblk + row) * K + kt + c8);
    }
#endif
    // --- B tile: 32x128 u16, store transposed [col][k] ---
#pragma unroll
    for (int i = 0; i < 2; ++i) {
      int v   = tid + i * 256;
      int row = v >> 4;
      int c8  = (v & 15) * 8;
      uint4 d = *(const uint4*)(Bm + (size_t)(kt + row) * N + nblk + c8);
      const unsigned short* dp = (const unsigned short*)&d;
#pragma unroll
      for (int e = 0; e < 8; ++e) Bs[(c8 + e) * 40 + row] = dp[e];
    }
    if (kt + 32 < K)   // hint next B tile toward the caches
      __builtin_prefetch(Bm + (size_t)(kt + 32 + (tid >> 4)) * N + nblk + (tid & 15) * 8, 0, 1);
#if HAVE_TDM
    __builtin_amdgcn_s_wait_tensorcnt(0);
#endif
    __syncthreads();

    // --- fragments (contiguous k = half*16 + e in A and B) + 8 WMMAs ---
    ushort16v af[2];
#pragma unroll
    for (int i = 0; i < 2; ++i) {
      const unsigned short* ap = &As[(wm * 32 + i * 16 + l15) * 40 + half * 16];
#pragma unroll
      for (int e = 0; e < 8; ++e) af[i][e] = ap[e];          // K = half*16 + e
#pragma unroll
      for (int e = 0; e < 8; ++e) af[i][8 + e] = ap[8 + e];  // K = half*16 + 8 + e
    }
#pragma unroll
    for (int jn = 0; jn < 4; ++jn) {
      ushort16v bf;
      const unsigned short* bp = &Bs[(wn * 64 + jn * 16 + l15) * 40 + half * 16];
#pragma unroll
      for (int e = 0; e < 16; ++e) bf[e] = bp[e];            // K = half*16 + e
#pragma unroll
      for (int i = 0; i < 2; ++i) acc[i][jn] = wmma_bf16(af[i], bf, acc[i][jn]);
    }
  }

  // --- epilogue: C layout row = r + half*8, col = l15 ---
#pragma unroll
  for (int i = 0; i < 2; ++i)
#pragma unroll
    for (int jn = 0; jn < 4; ++jn)
#pragma unroll
      for (int r = 0; r < 8; ++r) {
        int row = mblk + wm * 32 + i * 16 + r + half * 8;
        int col = nblk + wn * 64 + jn * 16 + l15;
        float vf = acc[i][jn][r] * alpha;
        if (STORE_BF16) ((unsigned short*)Cout)[(size_t)row * N + col] = f2bf(vf);
        else            ((float*)Cout)[(size_t)row * N + col] = vf;
      }
}

// ---------------------------------------------------------------------------
// Flash attention (causal, online softmax). Wave = 16 query rows; the 8 waves
// of a block share cooperatively staged K/V tiles (unified trip count, ~6%
// extra masked work vs per-wave bounds, in exchange for 8x less load traffic
// and vectorized LDS fragment reads with zero repacking VALU).
//   K tile: 32x64 row-major in LDS via TDM (padded stride 72 u16).
//   V tile: stored transposed [64][32+pad] so PV B-frags are contiguous.
// ---------------------------------------------------------------------------
__global__ void __launch_bounds__(256) flash_attn_kernel(
    const unsigned short* __restrict__ q,
    const unsigned short* __restrict__ kv,
    unsigned short* __restrict__ o_out) {
  constexpr int S = 2048, D = 1024, LDKV = 2048;
  __shared__ unsigned short Ks[32 * 72];        // 128B row + 16B pad (TDM 4/3)
  __shared__ unsigned short Vt[64 * 40];        // transposed V tile
  __shared__ unsigned short Pl[8][16 * 40];     // per-wave P transpose scratch

  const int tid  = threadIdx.x;
  const int lane = tid & 31;
  const int w    = tid >> 5;
  const int half = lane >> 4;
  const int l15  = lane & 15;
  const int bh   = blockIdx.y;
  const int b    = bh >> 4;          // H = 16
  const int h    = bh & 15;
  const int qb   = blockIdx.x * 128 + w * 16;   // this wave's first query row
  const size_t qrow0 = (size_t)(b * S + qb);

  // --- resident Q A-fragments (16x64 -> two 16x32 frags) ---
  ushort16v qa[2];
#pragma unroll
  for (int kc = 0; kc < 2; ++kc) {
    const unsigned short* qp = q + (qrow0 + l15) * D + h * 64 + kc * 32 + half * 16;
#pragma unroll
    for (int e = 0; e < 8; ++e) qa[kc][e] = qp[e];
#pragma unroll
    for (int e = 0; e < 8; ++e) qa[kc][8 + e] = qp[8 + e];
  }

  floatx8 oacc[4];
#pragma unroll
  for (int n4 = 0; n4 < 4; ++n4)
#pragma unroll
    for (int r = 0; r < 8; ++r) oacc[n4][r] = 0.0f;
  float mrow[8], lrow[8];
#pragma unroll
  for (int r = 0; r < 8; ++r) { mrow[r] = -__builtin_inff(); lrow[r] = 0.0f; }

  const unsigned short* Kg = kv + (size_t)(b * S) * LDKV + h * 64;
  const unsigned short* Vg = Kg + 1024;   // V half of the KV projection

  const int nbmax = (blockIdx.x + 1) * 4;       // unified causal trip count
  for (int j = 0; j < nbmax; ++j) {
    const int kj = j * 32;
    __syncthreads();   // previous tiles fully consumed
#if HAVE_TDM
    if (w == 0)        // K tile DMA: 32 rows x 64 bf16, stride LDKV
      tdm_load_2d((unsigned)(uintptr_t)Ks, Kg + (size_t)kj * LDKV,
                  /*w=*/64, /*h=*/32, (unsigned long long)LDKV,
                  /*interval 16x8B=128B*/ 4, /*pad 4 dw=16B*/ 3);
#else
    {
      int row = tid >> 3, c8 = (tid & 7) * 8;
      *(uint4*)&Ks[row * 72 + c8] =
          *(const uint4*)(Kg + (size_t)(kj + row) * LDKV + c8);
    }
#endif
    {  // V tile: coalesced b128 reads, transposed scatter into LDS
      int row = tid >> 3, c8 = (tid & 7) * 8;
      uint4 d = *(const uint4*)(Vg + (size_t)(kj + row) * LDKV + c8);
      const unsigned short* dp = (const unsigned short*)&d;
#pragma unroll
      for (int e = 0; e < 8; ++e) Vt[(c8 + e) * 40 + row] = dp[e];
    }
#if HAVE_TDM
    __builtin_amdgcn_s_wait_tensorcnt(0);
#endif
    __syncthreads();

    if (kj <= qb + 15) {   // uniform per wave: causally relevant block
      // --- scores: two 16x16 C tiles from LDS K (contiguous 32B per lane) ---
      floatx8 c[2];
#pragma unroll
      for (int t = 0; t < 2; ++t) {
        floatx8 s;
#pragma unroll
        for (int r = 0; r < 8; ++r) s[r] = 0.0f;
#pragma unroll
        for (int kc = 0; kc < 2; ++kc) {
          ushort16v bfrag;   // B[kk][n] = K[key n][kk]
          const unsigned short* kp = &Ks[(t * 16 + l15) * 72 + kc * 32 + half * 16];
#pragma unroll
          for (int e = 0; e < 16; ++e) bfrag[e] = kp[e];
          s = wmma_bf16(qa[kc], bfrag, s);
        }
        c[t] = s;
      }

      // --- causal mask (diagonal blocks only) ---
      if (kj + 31 > qb) {
#pragma unroll
        for (int t = 0; t < 2; ++t)
#pragma unroll
          for (int r = 0; r < 8; ++r) {
            int qi = qb + r + half * 8;
            int ki = kj + t * 16 + l15;
            if (ki > qi) c[t][r] = -50000.0f;
          }
      }

      // --- online softmax (row stats per (r, lane-half), 16-lane reductions) ---
      float p0[8], p1[8], esc[8];
#pragma unroll
      for (int r = 0; r < 8; ++r) {
        float mx = fmaxf(c[0][r], c[1][r]);
        mx = fmaxf(mx, __shfl_xor(mx, 1, 32));
        mx = fmaxf(mx, __shfl_xor(mx, 2, 32));
        mx = fmaxf(mx, __shfl_xor(mx, 4, 32));
        mx = fmaxf(mx, __shfl_xor(mx, 8, 32));
        float mn = fmaxf(mrow[r], mx);
        float sc = __expf(mrow[r] - mn);
        float a0 = __expf(c[0][r] - mn);
        float a1 = __expf(c[1][r] - mn);
        float ps = a0 + a1;
        ps += __shfl_xor(ps, 1, 32);
        ps += __shfl_xor(ps, 2, 32);
        ps += __shfl_xor(ps, 4, 32);
        ps += __shfl_xor(ps, 8, 32);
        lrow[r] = lrow[r] * sc + ps;
        mrow[r] = mn;
        esc[r] = sc; p0[r] = a0; p1[r] = a1;
      }

      // --- P: C-layout -> A-layout through per-wave LDS scratch ---
      unsigned short* pl = Pl[w];
#pragma unroll
      for (int r = 0; r < 8; ++r) {
        int m = r + half * 8;
        pl[m * 40 + l15]      = f2bf(p0[r]);
        pl[m * 40 + 16 + l15] = f2bf(p1[r]);
      }
      asm volatile("s_wait_dscnt 0" ::: "memory");  // wave-local LDS RAW fence
      ushort16v pa;
      {
        const unsigned short* pp = &pl[l15 * 40 + half * 16];
#pragma unroll
        for (int e = 0; e < 8; ++e) pa[e] = pp[e];
#pragma unroll
        for (int e = 0; e < 8; ++e) pa[8 + e] = pp[8 + e];
      }

      // --- O = O*esc + P @ V (B-frags contiguous from transposed V tile) ---
#pragma unroll
      for (int n4 = 0; n4 < 4; ++n4) {
#pragma unroll
        for (int r = 0; r < 8; ++r) oacc[n4][r] *= esc[r];
        ushort16v vfrag;   // B[kk][n] = V[key kj+kk][n]
        const unsigned short* vp = &Vt[(n4 * 16 + l15) * 40 + half * 16];
#pragma unroll
        for (int e = 0; e < 16; ++e) vfrag[e] = vp[e];
        oacc[n4] = wmma_bf16(pa, vfrag, oacc[n4]);
      }
    }
  }

  // --- normalize and store bf16 attention output [B*S, D] ---
  float inv[8];
#pragma unroll
  for (int r = 0; r < 8; ++r) inv[r] = 1.0f / lrow[r];
#pragma unroll
  for (int n4 = 0; n4 < 4; ++n4)
#pragma unroll
    for (int r = 0; r < 8; ++r) {
      size_t row = qrow0 + r + half * 8;
      int col = h * 64 + n4 * 16 + l15;
      o_out[row * D + col] = f2bf(oacc[n4][r] * inv[r]);
    }
}

// ---------------------------------------------------------------------------
// Launch: convert -> GEMM(Q) -> GEMM(KV) -> flash attention -> GEMM(Wo)
// ---------------------------------------------------------------------------
extern "C" void kernel_launch(void* const* d_in, const int* in_sizes, int n_in,
                              void* d_out, int out_size, void* d_ws, size_t ws_size,
                              hipStream_t stream) {
  (void)in_sizes; (void)n_in; (void)out_size; (void)ws_size;
  constexpr int B = 2, S = 2048, D = 1024;
  constexpr int BS = B * S;   // 4096 rows

  const float* x   = (const float*)d_in[0];
  const float* Wq  = (const float*)d_in[1];
  const float* Wkv = (const float*)d_in[2];
  const float* Wo  = (const float*)d_in[3];
  float* out = (float*)d_out;

  unsigned short* wsu = (unsigned short*)d_ws;
  size_t off = 0;
  unsigned short* x_bf   = wsu + off; off += (size_t)BS * D;      // 8 MB
  unsigned short* wq_bf  = wsu + off; off += (size_t)D * D;       // 2 MB
  unsigned short* wkv_bf = wsu + off; off += (size_t)D * 2 * D;   // 4 MB
  unsigned short* wo_bf  = wsu + off; off += (size_t)D * D;       // 2 MB
  unsigned short* q_bf   = wsu + off; off += (size_t)BS * D;      // 8 MB
  unsigned short* kv_bf  = wsu + off; off += (size_t)BS * 2 * D;  // 16 MB
  unsigned short* at_bf  = wsu + off; off += (size_t)BS * D;      // 8 MB

  f32_to_bf16_kernel<<<(BS * D + 255) / 256, 256, 0, stream>>>(x, x_bf, BS * D);
  f32_to_bf16_kernel<<<(D * D + 255) / 256, 256, 0, stream>>>(Wq, wq_bf, D * D);
  f32_to_bf16_kernel<<<(D * 2 * D + 255) / 256, 256, 0, stream>>>(Wkv, wkv_bf, D * 2 * D);
  f32_to_bf16_kernel<<<(D * D + 255) / 256, 256, 0, stream>>>(Wo, wo_bf, D * D);

  const float qscale = 0.125f;   // 1/sqrt(64), folded into the Q projection
  gemm_tile<true><<<dim3(D / 128, BS / 128), 256, 0, stream>>>(
      x_bf, wq_bf, q_bf, BS, D, D, qscale);
  gemm_tile<true><<<dim3(2 * D / 128, BS / 128), 256, 0, stream>>>(
      x_bf, wkv_bf, kv_bf, BS, 2 * D, D, 1.0f);
  flash_attn_kernel<<<dim3(S / 128, B * 16), 256, 0, stream>>>(q_bf, kv_bf, at_bf);
  gemm_tile<false><<<dim3(D / 128, BS / 128), 256, 0, stream>>>(
      at_bf, wo_bf, out, BS, D, D, 1.0f);
}
